// DecoderRNN_7395933684110
// MI455X (gfx1250) — matrix operand order repaired and checked by reading, not compiled, not run on hardware
//
#include <hip/hip_runtime.h>
#include <hip/hip_bf16.h>

// ---------------------------------------------------------------------------
// Problem constants (from reference)
// ---------------------------------------------------------------------------
#define EMBED   256
#define HIDDEN  512
#define VOCAB   32000
#define BATCH   64
#define TSTEPS  32

typedef __attribute__((ext_vector_type(16))) __bf16 v16bf;
typedef __attribute__((ext_vector_type(8)))  __bf16 v8bf;
typedef __attribute__((ext_vector_type(8)))  float  v8f;

// ---------------------------------------------------------------------------
// Helpers
// ---------------------------------------------------------------------------
__device__ __forceinline__ unsigned short f32_to_bf16_rne(float f) {
    unsigned int u = __float_as_uint(f);
    unsigned int r = u + 0x7FFFu + ((u >> 16) & 1u);   // round-to-nearest-even
    return (unsigned short)(r >> 16);
}

__device__ __forceinline__ float sigmoidf_fast(float x) {
    return 1.0f / (1.0f + __expf(-x));
}

// ---------------------------------------------------------------------------
// K0: f32 -> bf16 bulk convert (W_fc each launch; L2-resident afterwards)
// ---------------------------------------------------------------------------
__global__ void k_convert_bf16(const float* __restrict__ src,
                               unsigned short* __restrict__ dst, int n) {
    int i = blockIdx.x * blockDim.x + threadIdx.x;
    if (i < n) dst[i] = f32_to_bf16_rne(src[i]);
}

// ---------------------------------------------------------------------------
// K1: zero fill (h/c init)
// ---------------------------------------------------------------------------
__global__ void k_zero_f32(float* __restrict__ p, int n) {
    int i = blockIdx.x * blockDim.x + threadIdx.x;
    if (i < n) p[i] = 0.0f;
}

// ---------------------------------------------------------------------------
// K2: zero the last time slot out[:, T-1, :]
// ---------------------------------------------------------------------------
__global__ void k_zero_last_slot(float* __restrict__ out) {
    int v = blockIdx.x * blockDim.x + threadIdx.x;     // 0..VOCAB-1
    int b = blockIdx.y;                                // 0..BATCH-1
    out[(size_t)b * TSTEPS * VOCAB + (size_t)(TSTEPS - 1) * VOCAB + v] = 0.0f;
}

// ---------------------------------------------------------------------------
// K3: fused LSTM cell. One block per batch row (512 threads, 1 hidden unit
// each). x/h staged in LDS; weight rows read as float4 (b128 loads). Each
// thread computes its 4 gates (i,f,g,o), the nonlinearity, and writes h/c
// plus the bf16 copy of h for the WMMA FC.
// ---------------------------------------------------------------------------
__global__ void __launch_bounds__(HIDDEN)
k_lstm_cell(const float* __restrict__ x,      // [B, EMBED]
            const float* __restrict__ h_in,   // [B, HIDDEN]
            const float* __restrict__ c_in,   // [B, HIDDEN]
            const float* __restrict__ Wih,    // [4H, EMBED]
            const float* __restrict__ Whh,    // [4H, HIDDEN]
            const float* __restrict__ bih,    // [4H]
            const float* __restrict__ bhh,    // [4H]
            float* __restrict__ h_out,        // [B, HIDDEN]
            float* __restrict__ c_out,        // [B, HIDDEN]
            unsigned short* __restrict__ hbf_out) // [B, HIDDEN] bf16
{
    __shared__ float sx[EMBED];
    __shared__ float sh[HIDDEN];

    const int b = blockIdx.x;
    const int j = threadIdx.x;                 // 0..HIDDEN-1

    if (j < EMBED) sx[j] = x[b * EMBED + j];
    sh[j] = h_in[b * HIDDEN + j];
    __syncthreads();

    float gi = bih[0 * HIDDEN + j] + bhh[0 * HIDDEN + j];
    float gf = bih[1 * HIDDEN + j] + bhh[1 * HIDDEN + j];
    float gg = bih[2 * HIDDEN + j] + bhh[2 * HIDDEN + j];
    float go = bih[3 * HIDDEN + j] + bhh[3 * HIDDEN + j];

    const float4* wi0 = (const float4*)(Wih + (size_t)(0 * HIDDEN + j) * EMBED);
    const float4* wi1 = (const float4*)(Wih + (size_t)(1 * HIDDEN + j) * EMBED);
    const float4* wi2 = (const float4*)(Wih + (size_t)(2 * HIDDEN + j) * EMBED);
    const float4* wi3 = (const float4*)(Wih + (size_t)(3 * HIDDEN + j) * EMBED);
    #pragma unroll 2
    for (int e4 = 0; e4 < EMBED / 4; ++e4) {
        float4 w0 = wi0[e4], w1 = wi1[e4], w2 = wi2[e4], w3 = wi3[e4];
        float x0 = sx[4 * e4 + 0], x1 = sx[4 * e4 + 1];
        float x2 = sx[4 * e4 + 2], x3 = sx[4 * e4 + 3];
        gi = fmaf(x0, w0.x, fmaf(x1, w0.y, fmaf(x2, w0.z, fmaf(x3, w0.w, gi))));
        gf = fmaf(x0, w1.x, fmaf(x1, w1.y, fmaf(x2, w1.z, fmaf(x3, w1.w, gf))));
        gg = fmaf(x0, w2.x, fmaf(x1, w2.y, fmaf(x2, w2.z, fmaf(x3, w2.w, gg))));
        go = fmaf(x0, w3.x, fmaf(x1, w3.y, fmaf(x2, w3.z, fmaf(x3, w3.w, go))));
    }

    const float4* wh0 = (const float4*)(Whh + (size_t)(0 * HIDDEN + j) * HIDDEN);
    const float4* wh1 = (const float4*)(Whh + (size_t)(1 * HIDDEN + j) * HIDDEN);
    const float4* wh2 = (const float4*)(Whh + (size_t)(2 * HIDDEN + j) * HIDDEN);
    const float4* wh3 = (const float4*)(Whh + (size_t)(3 * HIDDEN + j) * HIDDEN);
    #pragma unroll 2
    for (int e4 = 0; e4 < HIDDEN / 4; ++e4) {
        float4 w0 = wh0[e4], w1 = wh1[e4], w2 = wh2[e4], w3 = wh3[e4];
        float h0 = sh[4 * e4 + 0], h1 = sh[4 * e4 + 1];
        float h2 = sh[4 * e4 + 2], h3 = sh[4 * e4 + 3];
        gi = fmaf(h0, w0.x, fmaf(h1, w0.y, fmaf(h2, w0.z, fmaf(h3, w0.w, gi))));
        gf = fmaf(h0, w1.x, fmaf(h1, w1.y, fmaf(h2, w1.z, fmaf(h3, w1.w, gf))));
        gg = fmaf(h0, w2.x, fmaf(h1, w2.y, fmaf(h2, w2.z, fmaf(h3, w2.w, gg))));
        go = fmaf(h0, w3.x, fmaf(h1, w3.y, fmaf(h2, w3.z, fmaf(h3, w3.w, go))));
    }

    float c_new = sigmoidf_fast(gf) * c_in[b * HIDDEN + j]
                + sigmoidf_fast(gi) * tanhf(gg);
    float h_new = sigmoidf_fast(go) * tanhf(c_new);

    c_out[b * HIDDEN + j]   = c_new;
    h_out[b * HIDDEN + j]   = h_new;
    hbf_out[b * HIDDEN + j] = f32_to_bf16_rne(h_new);
}

// ---------------------------------------------------------------------------
// K4: FC projection via v_wmma_f32_16x16x32_bf16.
// logits[64, 32000] = h[64,512] @ W_fc[32000,512]^T + b_fc.
//
// - The whole A matrix (h, 64x512 bf16 = 64 KB) is staged once per block
//   into LDS (320 KB/WGP on CDNA5) -> A fragments come from ds_load_b128,
//   eliminating the 2000x redundant global re-read of h.
// - One wave owns one N-tile (16 vocab columns) and iterates all 4 M-tiles,
//   reusing the B fragment; B rows are contiguous in W_fc (row-major [n][k])
//   and stream from L2 (32.75 MB bf16, fits the 192 MB L2).
// - B is double-buffered: chunk k+1's two b128 loads are issued before the
//   four WMMAs of chunk k, so s_wait_loadcnt never has to drain to zero.
//
// Fragment layouts per CDNA5 ISA 7.12.2 (wave32):
//   A (16x32 bf16): lane l (hi=l>>4, m=l&15) holds row m,
//                   K = {k0+hi*8 .. +7} then {k0+16+hi*8 .. +7}.
//   B (32x16 bf16): lane l holds column (l&15), K = {k0+hi*8 .. +7} then {k0+16+hi*8 .. +7}.
//   D (16x16 f32):  VGPR r = row mTile*16 + hi*8 + r, column nTile*16 + (l&15).
// ---------------------------------------------------------------------------
__global__ void __launch_bounds__(256)
k_fc_wmma(const __bf16* __restrict__ hbf,   // [64, 512] bf16
          const __bf16* __restrict__ wfc,   // [32000, 512] bf16
          const float*  __restrict__ bias,  // [32000]
          float* __restrict__ logits)       // [64, 32000]
{
    __shared__ __align__(16) unsigned short sA_us[BATCH * HIDDEN]; // 64 KB

    // Cooperative stage of the full A matrix into LDS (4096 x 16B).
    {
        const uint4* src = (const uint4*)hbf;
        uint4*       dst = (uint4*)sA_us;
        #pragma unroll
        for (int i = 0; i < (BATCH * HIDDEN * 2 / 16) / 256; ++i)
            dst[i * 256 + threadIdx.x] = src[i * 256 + threadIdx.x];
    }
    __syncthreads();
    const __bf16* sA = (const __bf16*)sA_us;

    const int lane  = threadIdx.x & 31;
    const int wave  = threadIdx.x >> 5;
    const int nTile = blockIdx.x * 8 + wave;          // 0..1999 (grid exact)
    const int l15   = lane & 15;
    const int hi    = lane >> 4;
    const int ncol  = nTile * 16 + l15;

    const __bf16* brow = wfc + (size_t)ncol * HIDDEN + hi * 8;
    __builtin_prefetch(brow, 0, 3);                   // global_prefetch_b8

    v8f acc[4] = {v8f{}, v8f{}, v8f{}, v8f{}};

    // B double buffer: current chunk in (bc0,bc1); next prefetched into (bn0,bn1).
    v8bf bc0 = *(const v8bf*)(brow);
    v8bf bc1 = *(const v8bf*)(brow + 16);

    for (int k0 = 0; k0 < HIDDEN; k0 += 32) {
        const int kn = (k0 + 32) & (HIDDEN - 1);      // wraps to 0 on last iter
        v8bf bn0 = *(const v8bf*)(brow + kn);
        v8bf bn1 = *(const v8bf*)(brow + kn + 16);

        v16bf bv;
        #pragma unroll
        for (int i = 0; i < 8; ++i) { bv[i] = bc0[i]; bv[i + 8] = bc1[i]; }

        const int abase = k0 + hi * 8;
        #pragma unroll
        for (int m = 0; m < 4; ++m) {
            const __bf16* arow = sA + (m * 16 + l15) * HIDDEN + abase;
            v8bf a0 = *(const v8bf*)(arow);           // ds_load_b128
            v8bf a1 = *(const v8bf*)(arow + 16);      // ds_load_b128
            v16bf av;
            #pragma unroll
            for (int i = 0; i < 8; ++i) { av[i] = a0[i]; av[i + 8] = a1[i]; }

            acc[m] = __builtin_amdgcn_wmma_f32_16x16x32_bf16(
                false, av, false, bv, (short)0, acc[m], false, false);
        }

        bc0 = bn0; bc1 = bn1;
    }

    const float bval = bias[ncol];
    #pragma unroll
    for (int m = 0; m < 4; ++m) {
        const int rbase = m * 16 + hi * 8;
        #pragma unroll
        for (int r = 0; r < 8; ++r) {
            logits[(size_t)(rbase + r) * VOCAB + ncol] = acc[m][r] + bval;
        }
    }
}

// ---------------------------------------------------------------------------
// K5: per-row softmax + first-index argmax over VOCAB. Block per batch row.
// If outp == nullptr only the argmax index is produced (feature step).
// outp points at out + t*VOCAB; batch stride is TSTEPS*VOCAB.
// ---------------------------------------------------------------------------
__global__ void __launch_bounds__(512)
k_softmax_argmax(const float* __restrict__ logits, // [64, VOCAB]
                 float* __restrict__ outp,
                 int* __restrict__ idx)
{
    __shared__ float smax[512];
    __shared__ int   simx[512];
    __shared__ float ssum[512];

    const int b = blockIdx.x;
    const int t = threadIdx.x;                // 512 threads
    const float* row = logits + (size_t)b * VOCAB;

    float best = -__builtin_inff();
    int   bidx = 0;
    for (int v = t; v < VOCAB; v += 512) {
        float xv = row[v];
        if (xv > best) { best = xv; bidx = v; }
    }
    smax[t] = best; simx[t] = bidx;
    __syncthreads();
    for (int s = 256; s > 0; s >>= 1) {
        if (t < s) {
            float o = smax[t + s]; int oi = simx[t + s];
            if (o > smax[t] || (o == smax[t] && oi < simx[t])) {
                smax[t] = o; simx[t] = oi;
            }
        }
        __syncthreads();
    }
    const float rmax = smax[0];
    if (t == 0) idx[b] = simx[0];

    float lsum = 0.0f;
    for (int v = t; v < VOCAB; v += 512) lsum += __expf(row[v] - rmax);
    ssum[t] = lsum;
    __syncthreads();
    for (int s = 256; s > 0; s >>= 1) {
        if (t < s) ssum[t] += ssum[t + s];
        __syncthreads();
    }
    const float inv = 1.0f / ssum[0];

    if (outp != nullptr) {
        float* orow = outp + (size_t)b * TSTEPS * VOCAB;
        for (int v = t; v < VOCAB; v += 512)
            orow[v] = __expf(row[v] - rmax) * inv;
    }
}

// ---------------------------------------------------------------------------
// K6: greedy embedding gather: x[b,:] = embed_table[idx[b], :]
// ---------------------------------------------------------------------------
__global__ void __launch_bounds__(EMBED)
k_embed(const float* __restrict__ table,   // [VOCAB, EMBED]
        const int* __restrict__ idx,       // [B]
        float* __restrict__ x)             // [B, EMBED]
{
    const int b = blockIdx.x;
    const int e = threadIdx.x;             // EMBED threads
    x[b * EMBED + e] = table[(size_t)idx[b] * EMBED + e];
}

// ---------------------------------------------------------------------------
// Host-side orchestration (graph-capture safe: kernels only, all on stream)
// ---------------------------------------------------------------------------
extern "C" void kernel_launch(void* const* d_in, const int* in_sizes, int n_in,
                              void* d_out, int out_size, void* d_ws, size_t ws_size,
                              hipStream_t stream) {
    (void)in_sizes; (void)n_in; (void)out_size; (void)ws_size;

    const float* features    = (const float*)d_in[0];
    /* d_in[1] captions (int64) unused: greedy decode ignores them */
    const float* embed_table = (const float*)d_in[2];
    const float* W_ih        = (const float*)d_in[3];
    const float* W_hh        = (const float*)d_in[4];
    const float* b_ih        = (const float*)d_in[5];
    const float* b_hh        = (const float*)d_in[6];
    const float* W_fc        = (const float*)d_in[7];
    const float* b_fc        = (const float*)d_in[8];
    float* out = (float*)d_out;

    // Workspace layout (256B aligned chunks)
    char* p = (char*)d_ws;
    unsigned short* wfcb = (unsigned short*)p; p += (size_t)VOCAB * HIDDEN * 2;   // 32.75 MB
    float* logits = (float*)p;                 p += (size_t)BATCH * VOCAB * 4;    //  8.19 MB
    float* h      = (float*)p;                 p += (size_t)BATCH * HIDDEN * 4;
    float* c      = (float*)p;                 p += (size_t)BATCH * HIDDEN * 4;
    float* htmp   = (float*)p;                 p += (size_t)BATCH * HIDDEN * 4;
    float* ctmp   = (float*)p;                 p += (size_t)BATCH * HIDDEN * 4;
    unsigned short* hbf = (unsigned short*)p;  p += (size_t)BATCH * HIDDEN * 2;
    float* x      = (float*)p;                 p += (size_t)BATCH * EMBED * 4;
    int*   idx    = (int*)p;                   p += 256;

    // Preprocessing
    {
        const int n = VOCAB * HIDDEN;                       // 16,384,000
        k_convert_bf16<<<(n + 255) / 256, 256, 0, stream>>>(W_fc, wfcb, n);
    }
    k_zero_f32<<<(BATCH * HIDDEN * 2 + 255) / 256, 256, 0, stream>>>(h, BATCH * HIDDEN * 2); // h,c contiguous
    k_zero_last_slot<<<dim3(VOCAB / 256, BATCH), 256, 0, stream>>>(out);

    // Feature step: state NOT carried (h,c stay zero); only x0 produced.
    k_lstm_cell<<<BATCH, HIDDEN, 0, stream>>>(features, h, c, W_ih, W_hh, b_ih, b_hh,
                                              htmp, ctmp, hbf);
    k_fc_wmma<<<(VOCAB / 16) / 8, 256, 0, stream>>>((const __bf16*)hbf,
                                                    (const __bf16*)wfcb, b_fc, logits);
    k_softmax_argmax<<<BATCH, 512, 0, stream>>>(logits, nullptr, idx);
    k_embed<<<BATCH, EMBED, 0, stream>>>(embed_table, idx, x);

    // T-1 greedy decode steps; step t writes out[:, t, :].
    for (int t = 0; t < TSTEPS - 1; ++t) {
        k_lstm_cell<<<BATCH, HIDDEN, 0, stream>>>(x, h, c, W_ih, W_hh, b_ih, b_hh,
                                                  h, c, hbf);
        k_fc_wmma<<<(VOCAB / 16) / 8, 256, 0, stream>>>((const __bf16*)hbf,
                                                        (const __bf16*)wfcb, b_fc, logits);
        k_softmax_argmax<<<BATCH, 512, 0, stream>>>(logits, out + (size_t)t * VOCAB, idx);
        if (t < TSTEPS - 2)
            k_embed<<<BATCH, EMBED, 0, stream>>>(embed_table, idx, x);
    }
}
